// CustomBiLSTM_65146063945743
// MI455X (gfx1250) — matrix operand order repaired and checked by reading, not compiled, not run on hardware
//
#include <hip/hip_runtime.h>
#include <hip/hip_bf16.h>
#include <math.h>

// ---------------------------------------------------------------------------
// Bidirectional 2-layer LSTM with Mamba-gated input gate, CDNA5 (gfx1250).
// Persistent-scan kernels with grid-wide sync; all GEMMs via
// v_wmma_f32_16x16x32_bf16; stage-G B-operand staged through LDS with
// GLOBAL_LOAD_ASYNC_TO_LDS_B128 double buffering (ASYNCcnt-tracked).
// ---------------------------------------------------------------------------

#define H_    256
#define B_    256
#define T_    128
#define DIN_  512
#define NWG   64
#define TPB   256
#define WAVES (TPB / 32)
#define LDSROW 40          // ushorts per staged LDS row (80B, bank-spread)

typedef __attribute__((ext_vector_type(16))) __bf16        v16bf;
typedef __attribute__((ext_vector_type(8)))  float         v8f;
typedef __attribute__((ext_vector_type(4)))  unsigned int  v4u;

#define V8F_ZERO {0.f,0.f,0.f,0.f,0.f,0.f,0.f,0.f}

union FragU { v16bf v; v4u q[2]; };

#if defined(__has_builtin)
#define HB(x) __has_builtin(x)
#else
#define HB(x) 0
#endif

// Builtin wants v4i pointers in AS(1) (printed "__device__") / AS(3).
typedef __attribute__((__vector_size__(16))) int vi4;
typedef vi4 __attribute__((address_space(1))) vi4_g;
typedef vi4 __attribute__((address_space(3))) vi4_l;

// 16B global -> LDS async copy (fire-and-forget, ASYNCcnt)
__device__ inline void async_b128(const void* g, void* l) {
#if HB(__builtin_amdgcn_global_load_async_to_lds_b128)
    __builtin_amdgcn_global_load_async_to_lds_b128((vi4_g*)g, (vi4_l*)l, 0, 0);
#else
    unsigned loff = (unsigned)(size_t)(__attribute__((address_space(3))) char*)l;
    asm volatile("global_load_async_to_lds_b128 %0, %1, off"
                 :: "v"(loff), "v"(g) : "memory");
#endif
}

__device__ inline void async_wait0() {
#if HB(__builtin_amdgcn_s_wait_asynccnt)
    __builtin_amdgcn_s_wait_asynccnt(0);
#else
    asm volatile("s_wait_asynccnt 0" ::: "memory");
#endif
}

__device__ inline unsigned short f2bf(float f) {
    unsigned u = __float_as_uint(f);
    u += 0x7FFFu + ((u >> 16) & 1u);           // round-to-nearest-even
    return (unsigned short)(u >> 16);
}

// A fragment: row-major M x K bf16, tile (row0, k0), K-slab of 32.
// lane<16 : row=lane,    elems = K[k0..k0+15]
// lane>=16: row=lane-16, elems = K[k0+16..k0+31]
__device__ inline v16bf ldA(const unsigned short* base, int lda,
                            int row0, int k0, int lane) {
    int half = lane >> 4, r = lane & 15;
    const unsigned short* p = base + (size_t)(row0 + r) * lda + k0 + half * 16;
    FragU u;
    u.q[0] = *reinterpret_cast<const v4u*>(p);
    u.q[1] = *reinterpret_cast<const v4u*>(p + 8);
    return u.v;
}

// B fragment from row-major weight W (N x K): B[k][n] = W[n][k].
// lane<16 : col=lane,    elems = W[col][k0..k0+15]
// lane>=16: col=lane-16, elems = W[col][k0+16..k0+31]
__device__ inline v16bf ldB(const unsigned short* w, int ldk,
                            int n0, int k0, int lane) {
    int half = lane >> 4, r = lane & 15;
    const unsigned short* p = w + (size_t)(n0 + r) * ldk + k0 + half * 16;
    FragU u;
    u.q[0] = *reinterpret_cast<const v4u*>(p);
    u.q[1] = *reinterpret_cast<const v4u*>(p + 8);
    return u.v;
}

// B fragment from an LDS-staged slab (16 rows x 32 K, row stride LDSROW)
__device__ inline v16bf ldB_lds(const unsigned short* base, int lane) {
    int half = lane >> 4, r = lane & 15;
    const unsigned short* p = base + r * LDSROW + half * 16;
    FragU u;
    u.q[0] = *reinterpret_cast<const v4u*>(p);
    u.q[1] = *reinterpret_cast<const v4u*>(p + 8);
    return u.v;
}

// Issue async copy of one B k-slab (16 rows x 32 K bf16 = 1KB) into LDS.
// 64 x 16B chunks over 32 lanes => 2 async b128 per lane.
__device__ inline void issueB(const unsigned short* w, int ldk, int n0, int k0,
                              unsigned short* buf, int lane) {
#pragma unroll
    for (int c = 0; c < 2; ++c) {
        int flat = lane + 32 * c;
        int row = flat >> 2, seg = flat & 3;
        async_b128(w + (size_t)(n0 + row) * ldk + k0 + seg * 8,
                   buf + row * LDSROW + seg * 8);
    }
}

__device__ inline v8f wmma_bf16(v16bf a, v16bf b, v8f c) {
    return __builtin_amdgcn_wmma_f32_16x16x32_bf16(
        /*neg_a=*/false, a, /*neg_b=*/false, b,
        /*c_mod=*/(short)0, c, /*reuse_a=*/false, /*reuse_b=*/false);
}

__device__ inline float sigm(float x)   { return 1.f / (1.f + __expf(-x)); }
__device__ inline float softpl(float x) { return x > 20.f ? x : log1pf(__expf(x)); }

// Device-wide barrier: counter/generation on device-scope atomics (L2).
__device__ inline void gsync(unsigned* cnt, unsigned* gen) {
    __threadfence();
    __syncthreads();
    if (threadIdx.x == 0) {
        unsigned g0   = atomicAdd(gen, 0u);
        unsigned prev = atomicAdd(cnt, 1u);
        if (prev == (unsigned)(gridDim.x - 1)) {
            atomicExch(cnt, 0u);
            atomicAdd(gen, 1u);
        } else {
            while (atomicAdd(gen, 0u) == g0) { __builtin_amdgcn_s_sleep(1); }
        }
    }
    __syncthreads();
    __threadfence();
}

// ---------------------------------------------------------------------------
__global__ void k_init(float* states, unsigned short* dt_bf, unsigned* sync) {
    int i = blockIdx.x * blockDim.x + threadIdx.x;
    int n = gridDim.x * blockDim.x;
    for (int j = i; j < 4 * B_ * H_; j += n) states[j] = 0.f;
    for (int j = i; j < B_ * 32;     j += n) dt_bf[j] = 0;
    if (i < 2) sync[i] = 0;
}

// ---------------------------------------------------------------------------
// Weight conversion f32 -> bf16. Wg[l][d] = combined [w_ih | w_hh] rows
// (1024 x 768, layer0 uses cols 0..383). Other mats stay row-major (N x K).
// ---------------------------------------------------------------------------
__global__ void k_convert(const float* __restrict__ wih0, const float* __restrict__ whh0,
                          const float* __restrict__ wih1, const float* __restrict__ whh1,
                          const float* __restrict__ inpj, const float* __restrict__ xpj,
                          const float* __restrict__ dtpj, const float* __restrict__ opj,
                          unsigned short* __restrict__ Wg,
                          unsigned short* __restrict__ inpj_bf,
                          unsigned short* __restrict__ xpj_bf,
                          unsigned short* __restrict__ dtpj_bf,
                          unsigned short* __restrict__ opj_bf) {
    const int NG = 4 * 1024 * 768;
    const int NI = 1024 * 256;
    const int NX = 48 * 512;
    const int ND = 512 * 32;
    const int NO = 256 * 512;
    const int total = NG + NI + NX + ND + NO;
    for (int i = blockIdx.x * blockDim.x + threadIdx.x; i < total;
         i += gridDim.x * blockDim.x) {
        if (i < NG) {
            int ld  = i / (1024 * 768);
            int rem = i - ld * (1024 * 768);
            int n = rem / 768, k = rem - n * 768;
            int l = ld >> 1, d = ld & 1;
            float v = 0.f;
            if (l == 0) {
                if (k < 128)      v = wih0[((size_t)d * 1024 + n) * 128 + k];
                else if (k < 384) v = whh0[((size_t)d * 1024 + n) * 256 + (k - 128)];
            } else {
                if (k < 512)      v = wih1[((size_t)d * 1024 + n) * 512 + k];
                else              v = whh1[((size_t)d * 1024 + n) * 256 + (k - 512)];
            }
            Wg[i] = f2bf(v);
        } else if (i < NG + NI) {
            int j = i - NG;           inpj_bf[j] = f2bf(inpj[j]);
        } else if (i < NG + NI + NX) {
            int j = i - NG - NI;      xpj_bf[j]  = f2bf(xpj[j]);
        } else if (i < NG + NI + NX + ND) {
            int j = i - NG - NI - NX;
            int n = j >> 5, k = j & 31;
            dtpj_bf[j] = (k < 16) ? f2bf(dtpj[n * 16 + k]) : (unsigned short)0;
        } else {
            int j = i - NG - NI - NX - ND; opj_bf[j] = f2bf(opj[j]);
        }
    }
}

// ---------------------------------------------------------------------------
// Persistent scan over T timesteps for one (layer, direction).
// Five WMMA stages per step, separated by grid syncs; elementwise fused in.
// ---------------------------------------------------------------------------
__global__ void __launch_bounds__(TPB) k_scan(
    int in_ch, int dir,
    const float* __restrict__ in_act, float* __restrict__ out_act,
    const unsigned short* __restrict__ Wg,
    const unsigned short* __restrict__ inpj_bf,
    const unsigned short* __restrict__ xpj_bf,
    const unsigned short* __restrict__ dtpj_bf,
    const unsigned short* __restrict__ opj_bf,
    const float* __restrict__ conv_w, const float* __restrict__ conv_b,
    const float* __restrict__ dt_b,   const float* __restrict__ Dm,
    const float* __restrict__ b_ih,   const float* __restrict__ b_hh,
    float* __restrict__ hS, float* __restrict__ cS,
    unsigned short* __restrict__ xt_bf,     // 2 x (256 x 768) ping-pong A-matrix
    unsigned short* __restrict__ u_bf,  float* __restrict__ gbuf,
    float* __restrict__ xm, unsigned short* __restrict__ xm_bf,
    float* __restrict__ sz,
    unsigned short* __restrict__ dt_bf, float* __restrict__ bc,
    unsigned short* __restrict__ y_bf,
    unsigned* __restrict__ scnt, unsigned* __restrict__ sgen) {

    // per-wave double-buffered B-slab staging (16 rows x 32 K, padded rows)
    __shared__ __align__(16) unsigned short ldsB[WAVES][2][16 * LDSROW];

    const int lane   = threadIdx.x & 31;
    const int wv     = threadIdx.x >> 5;
    const int gwave  = blockIdx.x * WAVES + wv;
    const int nwaves = gridDim.x * WAVES;
    const int gtid   = blockIdx.x * TPB + threadIdx.x;
    const int nthr   = gridDim.x * TPB;
    const int Kc     = in_ch + H_;
    const int kItG   = Kc >> 5;

    // ---- T0: build initial A-matrix [x_t0 | h] in bf16 (buffer 0) ----
    {
        int t0 = (dir == 0) ? 0 : (T_ - 1);
        for (int e = gtid; e < B_ * Kc; e += nthr) {
            int b = e / Kc, k = e - b * Kc;
            float v = (k < in_ch) ? in_act[((size_t)b * T_ + t0) * in_ch + k]
                                  : hS[b * H_ + (k - in_ch)];
            xt_bf[(size_t)b * 768 + k] = f2bf(v);
        }
    }
    gsync(scnt, sgen);

#pragma clang loop unroll(disable)
    for (int t = 0; t < T_; ++t) {
        const int tidx = (dir == 0) ? t : (T_ - 1 - t);
        const unsigned short* xt_cur = xt_bf + (size_t)(t & 1) * (B_ * 768);
        unsigned short*       xt_nxt = xt_bf + (size_t)((t + 1) & 1) * (B_ * 768);

        // ---- Stage G: g = [x_t|h] @ Wg^T (256x1024, K=Kc)
        //      B-operand async-staged through LDS, double buffered ----
        for (int idx = gwave; idx < 16 * 64; idx += nwaves) {
            int m0 = (idx >> 6) << 4, n0 = (idx & 63) << 4;
            v8f acc = V8F_ZERO;
            issueB(Wg, 768, n0, 0, &ldsB[wv][0][0], lane);       // slab 0
#pragma clang loop unroll(disable)
            for (int kk = 0; kk < kItG; ++kk) {
                int cur = kk & 1;
                async_wait0();                                   // slab kk resident
                v16bf b = ldB_lds(&ldsB[wv][cur][0], lane);      // ds_load_b128 x2
                v16bf a = ldA(xt_cur, 768, m0, kk * 32, lane);
                if (kk + 1 < kItG)                               // overlap next DMA
                    issueB(Wg, 768, n0, (kk + 1) * 32, &ldsB[wv][cur ^ 1][0], lane);
                acc = wmma_bf16(a, b, acc);
            }
            int half = lane >> 4, nl = lane & 15;
#pragma unroll
            for (int r = 0; r < 8; ++r) {
                int m = m0 + r + half * 8, n = n0 + nl;
                float v = acc[r];
                gbuf[(size_t)m * 1024 + n] = v;
                if (n0 < H_) u_bf[(size_t)m * H_ + n] = f2bf(v);  // mamba input
            }
        }
        gsync(scnt, sgen);

        // ---- Stage XZ: xz = u @ in_proj^T (256x1024, K=256); fuse conv+silu ----
        for (int idx = gwave; idx < 16 * 64; idx += nwaves) {
            int m0 = (idx >> 6) << 4, n0 = (idx & 63) << 4;
            v8f acc = V8F_ZERO;
            for (int kk = 0; kk < 8; ++kk) {
                v16bf a = ldA(u_bf,    H_, m0, kk * 32, lane);
                v16bf b = ldB(inpj_bf, H_, n0, kk * 32, lane);
                acc = wmma_bf16(a, b, acc);
            }
            int half = lane >> 4, nl = lane & 15;
#pragma unroll
            for (int r = 0; r < 8; ++r) {
                int m = m0 + r + half * 8, j = n0 + nl;
                float v = acc[r];
                if (j < DIN_) {
                    float pre = v * conv_w[j * 2 + 1] + conv_b[j];
                    float xv  = pre * sigm(pre);               // silu
                    xm  [(size_t)m * DIN_ + j] = xv;
                    xm_bf[(size_t)m * DIN_ + j] = f2bf(xv);
                } else {
                    int jz = j - DIN_;
                    sz[(size_t)m * DIN_ + jz] = v * sigm(v);   // silu(z)
                }
            }
        }
        gsync(scnt, sgen);

        // ---- Stage XDBL: x_dbl = xm @ x_proj^T (256x48, K=512); fuse dt/bc ----
        for (int idx = gwave; idx < 16; idx += nwaves) {
            int m0 = idx << 4;
            v8f a0 = V8F_ZERO, a1 = V8F_ZERO, a2 = V8F_ZERO;
            for (int kk = 0; kk < 16; ++kk) {
                v16bf a = ldA(xm_bf, DIN_, m0, kk * 32, lane);
                a0 = wmma_bf16(a, ldB(xpj_bf, DIN_, 0,  kk * 32, lane), a0);
                a1 = wmma_bf16(a, ldB(xpj_bf, DIN_, 16, kk * 32, lane), a1);
                a2 = wmma_bf16(a, ldB(xpj_bf, DIN_, 32, kk * 32, lane), a2);
            }
            int half = lane >> 4, nl = lane & 15;
#pragma unroll
            for (int r = 0; r < 8; ++r) {
                int m = m0 + r + half * 8;
                dt_bf[m * 32 + nl] = f2bf(a0[r]);              // dt tile (K-padded)
                float prod = a1[r] * a2[r];                    // Bm*Cm
#pragma unroll
                for (int o = 1; o < 16; o <<= 1) prod += __shfl_xor(prod, o, 32);
                if (nl == 0) bc[m] = prod;                     // row-sum over 16 cols
            }
        }
        gsync(scnt, sgen);

        // ---- Stage DELTA: delta_in = dt @ dt_proj^T (256x512, K=32 padded);
        //      fuse softplus + y; also prep next x_t (independent work) ----
        for (int idx = gwave; idx < 16 * 32; idx += nwaves) {
            int m0 = (idx >> 5) << 4, n0 = (idx & 31) << 4;
            v16bf a = ldA(dt_bf,   32, m0, 0, lane);
            v16bf b = ldB(dtpj_bf, 32, n0, 0, lane);
            v8f acc = V8F_ZERO;
            acc = wmma_bf16(a, b, acc);
            int half = lane >> 4, nl = lane & 15;
#pragma unroll
            for (int r = 0; r < 8; ++r) {
                int m = m0 + r + half * 8, j = n0 + nl;
                float delta = softpl(acc[r] + dt_b[j]);
                float xv = xm[(size_t)m * DIN_ + j];
                float y  = delta * xv * bc[m] + Dm[j] * xv;
                y *= sz[(size_t)m * DIN_ + j];
                y_bf[(size_t)m * DIN_ + j] = f2bf(y);
            }
        }
        if (t + 1 < T_) {
            int tn = (dir == 0) ? (t + 1) : (T_ - 2 - t);
            int tp = (dir == 0) ? ((t + 2 < T_) ? t + 2 : tn)
                                : ((T_ - 3 - t >= 0) ? T_ - 3 - t : tn);
            for (int e = gtid; e < B_ * in_ch; e += nthr) {
                int b = e / in_ch, i2 = e - b * in_ch;
                xt_nxt[(size_t)b * 768 + i2] =
                    f2bf(in_act[((size_t)b * T_ + tn) * in_ch + i2]);
                // warm L2 one extra step ahead of the recurrence
                __builtin_prefetch(&in_act[((size_t)b * T_ + tp) * in_ch + i2], 0, 1);
            }
        }
        gsync(scnt, sgen);

        // ---- Stage Y: mam = y @ out_proj^T (256x256, K=512);
        //      fuse LSTM gates + cell update + output + next-step h(bf16) ----
        for (int idx = gwave; idx < 16 * 16; idx += nwaves) {
            int m0 = (idx >> 4) << 4, n0 = (idx & 15) << 4;
            v8f acc = V8F_ZERO;
            for (int kk = 0; kk < 16; ++kk) {
                v16bf a = ldA(y_bf,   DIN_, m0, kk * 32, lane);
                v16bf b = ldB(opj_bf, DIN_, n0, kk * 32, lane);
                acc = wmma_bf16(a, b, acc);
            }
            int half = lane >> 4, nl = lane & 15;
#pragma unroll
            for (int r = 0; r < 8; ++r) {
                int m = m0 + r + half * 8, j = n0 + nl;
                float it = sigm(acc[r] + b_ih[j] + b_hh[j]);
                float ft = sigm(gbuf[(size_t)m * 1024 + 256 + j] + b_ih[256 + j] + b_hh[256 + j]);
                float gg = tanhf(gbuf[(size_t)m * 1024 + 512 + j] + b_ih[512 + j] + b_hh[512 + j]);
                float ot = sigm(gbuf[(size_t)m * 1024 + 768 + j] + b_ih[768 + j] + b_hh[768 + j]);
                float c  = ft * cS[m * H_ + j] + it * gg;
                float h  = ot * tanhf(c);
                cS[m * H_ + j] = c;
                hS[m * H_ + j] = h;
                out_act[((size_t)m * T_ + tidx) * (2 * H_) + dir * H_ + j] = h;
                xt_nxt[(size_t)m * 768 + in_ch + j] = f2bf(h);
            }
        }
        gsync(scnt, sgen);
    }
}

// ---------------------------------------------------------------------------
extern "C" void kernel_launch(void* const* d_in, const int* in_sizes, int n_in,
                              void* d_out, int out_size, void* d_ws, size_t ws_size,
                              hipStream_t stream) {
    (void)in_sizes; (void)n_in; (void)out_size; (void)ws_size;

    const float* x     = (const float*)d_in[0];
    const float* wih0  = (const float*)d_in[1];
    const float* whh0  = (const float*)d_in[2];
    const float* bih0  = (const float*)d_in[3];
    const float* bhh0  = (const float*)d_in[4];
    const float* wih1  = (const float*)d_in[5];
    const float* whh1  = (const float*)d_in[6];
    const float* bih1  = (const float*)d_in[7];
    const float* bhh1  = (const float*)d_in[8];
    const float* inpj  = (const float*)d_in[9];
    const float* convw = (const float*)d_in[10];
    const float* convb = (const float*)d_in[11];
    const float* xpj   = (const float*)d_in[12];
    const float* dtpj  = (const float*)d_in[13];
    const float* dtb   = (const float*)d_in[14];
    const float* Dm    = (const float*)d_in[15];
    const float* opj   = (const float*)d_in[16];
    float* out = (float*)d_out;

    char*  base = (char*)d_ws;
    size_t off  = 0;
    auto carve = [&](size_t bytes) -> void* {
        off = (off + 255) & ~(size_t)255;
        void* p = base + off;
        off += bytes;
        return p;
    };

    float* states = (float*)carve((size_t)4 * B_ * H_ * sizeof(float));
    float* hfw = states;
    float* cfw = states + 1 * B_ * H_;
    float* hbw = states + 2 * B_ * H_;
    float* cbw = states + 3 * B_ * H_;
    float* gbuf = (float*)carve((size_t)B_ * 1024 * sizeof(float));
    float* xm   = (float*)carve((size_t)B_ * DIN_ * sizeof(float));
    float* sz   = (float*)carve((size_t)B_ * DIN_ * sizeof(float));
    float* bc   = (float*)carve((size_t)B_ * sizeof(float));
    unsigned short* Wg      = (unsigned short*)carve((size_t)4 * 1024 * 768 * 2);
    unsigned short* inpj_bf = (unsigned short*)carve((size_t)1024 * 256 * 2);
    unsigned short* xpj_bf  = (unsigned short*)carve((size_t)48 * 512 * 2);
    unsigned short* dtpj_bf = (unsigned short*)carve((size_t)512 * 32 * 2);
    unsigned short* opj_bf  = (unsigned short*)carve((size_t)256 * 512 * 2);
    unsigned short* u_bf    = (unsigned short*)carve((size_t)B_ * H_ * 2);
    unsigned short* xm_bf   = (unsigned short*)carve((size_t)B_ * DIN_ * 2);
    unsigned short* y_bf    = (unsigned short*)carve((size_t)B_ * DIN_ * 2);
    unsigned short* dt_bf   = (unsigned short*)carve((size_t)B_ * 32 * 2);
    unsigned short* xt_bf   = (unsigned short*)carve((size_t)2 * B_ * 768 * 2);
    unsigned*       sync    = (unsigned*)carve(2 * sizeof(unsigned));
    float* buf0 = (float*)carve((size_t)B_ * T_ * 512 * sizeof(float));

    k_init<<<512, 256, 0, stream>>>(states, dt_bf, sync);
    k_convert<<<4096, 256, 0, stream>>>(wih0, whh0, wih1, whh1, inpj, xpj, dtpj, opj,
                                        Wg, inpj_bf, xpj_bf, dtpj_bf, opj_bf);

    // layer 0 forward / backward  (input x, 128ch) -> buf0
    k_scan<<<NWG, TPB, 0, stream>>>(128, 0, x, buf0, Wg + (size_t)0 * 786432,
        inpj_bf, xpj_bf, dtpj_bf, opj_bf, convw, convb, dtb, Dm,
        bih0, bhh0, hfw, cfw, xt_bf, u_bf, gbuf, xm, xm_bf, sz, dt_bf, bc, y_bf,
        sync, sync + 1);
    k_scan<<<NWG, TPB, 0, stream>>>(128, 1, x, buf0, Wg + (size_t)1 * 786432,
        inpj_bf, xpj_bf, dtpj_bf, opj_bf, convw, convb, dtb, Dm,
        bih0 + 1024, bhh0 + 1024, hbw, cbw, xt_bf, u_bf, gbuf, xm, xm_bf, sz, dt_bf, bc, y_bf,
        sync, sync + 1);

    // layer 1 forward / backward  (input buf0, 512ch) -> d_out
    k_scan<<<NWG, TPB, 0, stream>>>(512, 0, buf0, out, Wg + (size_t)2 * 786432,
        inpj_bf, xpj_bf, dtpj_bf, opj_bf, convw, convb, dtb, Dm,
        bih1, bhh1, hfw, cfw, xt_bf, u_bf, gbuf, xm, xm_bf, sz, dt_bf, bc, y_bf,
        sync, sync + 1);
    k_scan<<<NWG, TPB, 0, stream>>>(512, 1, buf0, out, Wg + (size_t)3 * 786432,
        inpj_bf, xpj_bf, dtpj_bf, opj_bf, convw, convb, dtb, Dm,
        bih1 + 1024, bhh1 + 1024, hbw, cbw, xt_bf, u_bf, gbuf, xm, xm_bf, sz, dt_bf, bc, y_bf,
        sync, sync + 1);
}
